// Decoder_84198538871328
// MI455X (gfx1250) — matrix operand order repaired and checked by reading, not compiled, not run on hardware
//
#include <hip/hip_runtime.h>
#include <hip/hip_bf16.h>
#include <math.h>

#define BSZ 8
#define NF 64
#define LATD 512
#define PT 45376
#define OFF_B1 12288
#define OFF_W2 12352
#define OFF_B2 20544
#define OFF_WS 20672
#define OFF_BS 45248
#define FIN 192
#define FH 64
#define FOUT 128
#define ZLD 200
#define HLD 72

typedef __attribute__((ext_vector_type(16))) _Float16 v16h;
typedef __attribute__((ext_vector_type(8)))  float    v8f;

// ---------------- WMMA fragment helpers (A and B share contiguous k order: k = 16*(l>>4) + j) ----

__device__ __forceinline__ v8f wmma_f16(v16h a, v16h b, v8f c) {
  return __builtin_amdgcn_wmma_f32_16x16x32_f16(false, a, false, b, (short)0, c,
                                                false, false);
}

// A: 16x32 f16, row-major weight matrix W (M x ldk). lane l holds M = mbase+(l&15);
// element j: K = kbase + 16*lg + j, lg = l>>4.
__device__ __forceinline__ v16h load_a_frag(const _Float16* __restrict__ W, int ldk,
                                            int mbase, int kbase, int lane) {
  const int m  = mbase + (lane & 15);
  const int lg = lane >> 4;
  const _Float16* row = W + (size_t)m * ldk + kbase + lg * 16;
  v16h a;
#pragma unroll
  for (int t = 0; t < 4; ++t) {
    a[2 * t]     = row[2 * t];
    a[2 * t + 1] = row[2 * t + 1];
    a[8 + 2 * t]     = row[8 + 2 * t];
    a[8 + 2 * t + 1] = row[8 + 2 * t + 1];
  }
  return a;
}

// B: 32x16 f16 from LDS tile stored [pixel][k] with row stride ld halves.
// lane l holds N(pixel) = nbase+(l&15); element j: K = kbase + 16*(l>>4) + j.
__device__ __forceinline__ v16h load_b_frag(const _Float16* __restrict__ T, int ld,
                                            int nbase, int kbase, int lane) {
  const int pix = nbase + (lane & 15);
  const _Float16* row = T + (size_t)pix * ld + kbase + (lane >> 4) * 16;
  v16h b;
#pragma unroll
  for (int j = 0; j < 16; ++j) b[j] = row[j];
  return b;
}

// ---------------- small helpers ----------------

__device__ __forceinline__ float ldz(const float* __restrict__ ch, int H, int W,
                                     int y, int x) {
  return (y < 0 || y >= H || x < 0 || x >= W) ? 0.f : ch[y * W + x];
}

// ---------------- hypernetwork: p[b][j] = sum_k lat[b][k] * hw[k][j] + hb[j] -------

__global__ void __launch_bounds__(256) hyper_kernel(const float* __restrict__ lat,
                                                    const float* __restrict__ hw,
                                                    const float* __restrict__ hb,
                                                    float* __restrict__ pbuf) {
  __shared__ float slat[BSZ * LATD];
  for (int i = threadIdx.x; i < BSZ * LATD; i += 256) slat[i] = lat[i];
  __syncthreads();
  int j = blockIdx.x * 256 + threadIdx.x;
  if (j >= PT) return;
  float acc[BSZ];
  float hbv = hb[j];
#pragma unroll
  for (int b = 0; b < BSZ; ++b) acc[b] = hbv;
  for (int k = 0; k < LATD; ++k) {
    float w = hw[(size_t)k * PT + j];
#pragma unroll
    for (int b = 0; b < BSZ; ++b) acc[b] += slat[b * LATD + k] * w;
  }
#pragma unroll
  for (int b = 0; b < BSZ; ++b) pbuf[(size_t)b * PT + j] = acc[b];
}

// ---------------- convert hyper params to f16 weight matrices ----------------------

__global__ void convert_kernel(const float* __restrict__ pbuf,
                               _Float16* __restrict__ w1h,
                               _Float16* __restrict__ w2h,
                               _Float16* __restrict__ wsh) {
  const int total = BSZ * (FH * FIN + FOUT * FH + FOUT * FIN);  // 45056 per batch
  int idx = blockIdx.x * blockDim.x + threadIdx.x;
  if (idx >= total) return;
  int b = idx / 45056;
  int i = idx - b * 45056;
  const float* pb_ = pbuf + (size_t)b * PT;
  if (i < FH * FIN) {
    w1h[(size_t)b * FH * FIN + i] = (_Float16)pb_[i];
  } else if (i < FH * FIN + FOUT * FH) {
    int q = i - FH * FIN;
    w2h[(size_t)b * FOUT * FH + q] = (_Float16)pb_[OFF_W2 + q];
  } else {
    int q = i - (FH * FIN + FOUT * FH);
    wsh[(size_t)b * FOUT * FIN + q] = (_Float16)pb_[OFF_WS + q];
  }
}

// ---------------- 3x3 gaussian blur (zero padding, cross-correlation) --------------

__global__ void blur_kernel(const float* __restrict__ in, float* __restrict__ outbuf,
                            int BC, int H, int W) {
  int total = BC * H * W;
  int idx = blockIdx.x * blockDim.x + threadIdx.x;
  if (idx >= total) return;
  int bc = idx / (H * W);
  int rem = idx - bc * H * W;
  int y = rem / W, x = rem - y * W;
  const float* ch = in + (size_t)bc * H * W;
  const float a = 0.60653065971263342f;
  const float inv = 1.f / ((1.f + 2.f * a) * (1.f + 2.f * a));
  const float w3[3] = {a, 1.f, a};
  float acc = 0.f;
#pragma unroll
  for (int dy = 0; dy < 3; ++dy)
#pragma unroll
    for (int dx = 0; dx < 3; ++dx)
      acc += w3[dy] * w3[dx] * ldz(ch, H, W, y + dy - 1, x + dx - 1);
  outbuf[idx] = acc * inv;
}

// ---------------- bilinear 2x upsample (half-pixel centers, edge clamp) ------------

__global__ void up2_kernel(const float* __restrict__ in, float* __restrict__ outbuf,
                           int H, int W) {
  int H2 = 2 * H, W2 = 2 * W;
  int total = BSZ * NF * H2 * W2;
  int idx = blockIdx.x * blockDim.x + threadIdx.x;
  if (idx >= total) return;
  int bc = idx / (H2 * W2);
  int rem = idx - bc * H2 * W2;
  int y2 = rem / W2, x2 = rem - y2 * W2;
  const float* src = in + (size_t)bc * H * W;
  int xa = x2 >> 1, ya = y2 >> 1;
  int xb = (x2 & 1) ? min(xa + 1, W - 1) : max(xa - 1, 0);
  int yb = (y2 & 1) ? min(ya + 1, H - 1) : max(ya - 1, 0);
  float vA = 0.75f * src[ya * W + xa] + 0.25f * src[ya * W + xb];
  float vB = 0.75f * src[yb * W + xa] + 0.25f * src[yb * W + xb];
  outbuf[idx] = 0.75f * vA + 0.25f * vB;
}

// ---------------- per-(b,cin): sobel features + instance-norm statistics -----------

__global__ void __launch_bounds__(256) stats_kernel(const float* __restrict__ out,
                                                    int H, int W,
                                                    float* __restrict__ meanp,
                                                    float* __restrict__ rstdp) {
  const int cin = blockIdx.x, b = blockIdx.y;
  const int HW = H * W;
  const float* ch = out + (size_t)(b * NF + cin) * HW;
  float s0 = 0, q0 = 0, s1 = 0, q1 = 0, s2 = 0, q2 = 0;
  for (int pi = threadIdx.x; pi < HW; pi += 256) {
    int y = pi / W, x = pi - y * W;
    float v = ch[pi];
    float a00 = ldz(ch, H, W, y - 1, x - 1), a01 = ldz(ch, H, W, y - 1, x);
    float a02 = ldz(ch, H, W, y - 1, x + 1);
    float a10 = ldz(ch, H, W, y, x - 1), a12 = ldz(ch, H, W, y, x + 1);
    float a20 = ldz(ch, H, W, y + 1, x - 1), a21 = ldz(ch, H, W, y + 1, x);
    float a22 = ldz(ch, H, W, y + 1, x + 1);
    float sx = (-a00 + a02 - 2.f * a10 + 2.f * a12 - a20 + a22) * 0.125f;
    float sy = (-a00 - 2.f * a01 - a02 + a20 + 2.f * a21 + a22) * 0.125f;
    s0 += v;  q0 += v * v;
    s1 += sx; q1 += sx * sx;
    s2 += sy; q2 += sy * sy;
  }
  __shared__ float part[8][6];
  float vals[6] = {s0, q0, s1, q1, s2, q2};
  int lane = threadIdx.x & 31, wv = threadIdx.x >> 5;
#pragma unroll
  for (int q = 0; q < 6; ++q) {
    float v = vals[q];
    for (int off = 16; off > 0; off >>= 1) v += __shfl_xor(v, off, 32);
    if (lane == 0) part[wv][q] = v;
  }
  __syncthreads();
  if (threadIdx.x == 0) {
    float tot[6] = {0, 0, 0, 0, 0, 0};
    for (int w = 0; w < 8; ++w)
#pragma unroll
      for (int q = 0; q < 6; ++q) tot[q] += part[w][q];
    float inv = 1.f / (float)HW;
#pragma unroll
    for (int t = 0; t < 3; ++t) {
      float m = tot[2 * t] * inv;
      float var = tot[2 * t + 1] * inv - m * m;
      meanp[b * FIN + cin + t * NF] = m;
      rstdp[b * FIN + cin + t * NF] = rsqrtf(var + 1e-5f);
    }
  }
}

// ---------------- write normalized z as f16, layout zbuf[b][pix][192] --------------

__global__ void __launch_bounds__(256) zwrite_kernel(const float* __restrict__ out,
                                                     const float* __restrict__ meanp,
                                                     const float* __restrict__ rstdp,
                                                     _Float16* __restrict__ zbuf,
                                                     int H, int W) {
  const int cin = blockIdx.x, b = blockIdx.y;
  const int HW = H * W;
  const float* ch = out + (size_t)(b * NF + cin) * HW;
  float m0 = meanp[b * FIN + cin],        r0 = rstdp[b * FIN + cin];
  float m1 = meanp[b * FIN + cin + 64],   r1 = rstdp[b * FIN + cin + 64];
  float m2 = meanp[b * FIN + cin + 128],  r2 = rstdp[b * FIN + cin + 128];
  _Float16* zb = zbuf + (size_t)b * HW * FIN;
  for (int pi = threadIdx.x; pi < HW; pi += 256) {
    int y = pi / W, x = pi - y * W;
    float v = ch[pi];
    float a00 = ldz(ch, H, W, y - 1, x - 1), a01 = ldz(ch, H, W, y - 1, x);
    float a02 = ldz(ch, H, W, y - 1, x + 1);
    float a10 = ldz(ch, H, W, y, x - 1), a12 = ldz(ch, H, W, y, x + 1);
    float a20 = ldz(ch, H, W, y + 1, x - 1), a21 = ldz(ch, H, W, y + 1, x);
    float a22 = ldz(ch, H, W, y + 1, x + 1);
    float sx = (-a00 + a02 - 2.f * a10 + 2.f * a12 - a20 + a22) * 0.125f;
    float sy = (-a00 - 2.f * a01 - a02 + a20 + 2.f * a21 + a22) * 0.125f;
    _Float16* row = zb + (size_t)pi * FIN;
    row[cin]       = (_Float16)((v - m0) * r0);
    row[cin + 64]  = (_Float16)((sx - m1) * r1);
    row[cin + 128] = (_Float16)((sy - m2) * r2);
  }
}

// ---------------- fused dyna_block + gate + residual (WMMA core) -------------------
// One block = one batch x 64-pixel tile. 8 waves.
//   stage:  z tile global -> LDS via GLOBAL_LOAD_ASYNC_TO_LDS_B128 (ASYNCcnt path)
//   phase1: h = relu(W1(64x192) @ z + b1)              -> LDS ht[pix][64] f16
//   phase2: s+y = Ws(128x192)@z + W2(128x64)@h + b2+bs ; out += lf * z0 * sigmoid(g)

__global__ void __launch_bounds__(256)
dyna_kernel(const _Float16* __restrict__ zbuf, const _Float16* __restrict__ w1h,
            const _Float16* __restrict__ w2h, const _Float16* __restrict__ wsh,
            const float* __restrict__ pbuf, const float* __restrict__ lfp,
            float* __restrict__ outp, int HW) {
  __shared__ _Float16 zt[64][ZLD];
  __shared__ _Float16 ht[64][HLD];
  const int b = blockIdx.y;
  const int pb = blockIdx.x;
  const int tid = threadIdx.x;
  const int lane = tid & 31, wv = tid >> 5, lg = (tid >> 4) & 1;

  // stage 64x192 f16 z tile straight into LDS: async copy, no VGPR round trip.
  // Flat shared pointers carry the LDS byte offset in their low 32 bits
  // (ISA 10.2 aperture mapping), which is exactly the VDST operand of
  // GLOBAL_LOAD_ASYNC_TO_LDS. Each lane moves 16B; 6 ops/thread = 24576B tile.
  {
    const int pix = tid >> 2, part = tid & 3;
    const _Float16* gsrc =
        zbuf + ((size_t)b * HW + (size_t)pb * 64 + pix) * FIN + part * 48;
    unsigned lds_off = (unsigned)(uintptr_t)(&zt[pix][part * 48]);
#pragma unroll
    for (int i = 0; i < 6; ++i) {
      const _Float16* gp = gsrc + i * 8;           // +16 bytes per step
      unsigned lo = lds_off + i * 16;
      asm volatile("global_load_async_to_lds_b128 %0, %1, off"
                   :: "v"(lo), "v"(gp) : "memory");
    }
    asm volatile("s_wait_asynccnt 0x0" ::: "memory");
  }
  __syncthreads();

  const float* pb_ = pbuf + (size_t)b * PT;

  // ---- phase 1: h (64 ch x 64 pix) = 16 tiles, 2 per wave, K=192 -> 6 wmma each
  const _Float16* W1b = w1h + (size_t)b * (FH * FIN);
#pragma unroll
  for (int t2 = 0; t2 < 2; ++t2) {
    int job = wv + 8 * t2;
    int mi = job >> 2, ni = job & 3;
    v8f acc;
#pragma unroll
    for (int r = 0; r < 8; ++r) acc[r] = pb_[OFF_B1 + mi * 16 + r + 8 * lg];
#pragma unroll
    for (int kk = 0; kk < 6; ++kk) {
      v16h a = load_a_frag(W1b, FIN, mi * 16, kk * 32, lane);
      v16h bb = load_b_frag(&zt[0][0], ZLD, ni * 16, kk * 32, lane);
      acc = wmma_f16(a, bb, acc);
    }
    int pixl = ni * 16 + (lane & 15);
#pragma unroll
    for (int r = 0; r < 8; ++r) {
      int chn = mi * 16 + r + 8 * lg;
      float v = acc[r] > 0.f ? acc[r] : 0.f;
      ht[pixl][chn] = (_Float16)v;
    }
  }
  __syncthreads();

  // ---- phase 2: paired (z rows m, gate rows m+64) tiles; fused gate + residual
  float lf = *lfp;
  lf = fminf(fmaxf(lf, 0.001f), 1000.f);
  const _Float16* Wsb = wsh + (size_t)b * (FOUT * FIN);
  const _Float16* W2b = w2h + (size_t)b * (FOUT * FH);
#pragma unroll
  for (int t2 = 0; t2 < 2; ++t2) {
    int job = wv + 8 * t2;
    int mp = job >> 2, ni = job & 3;
    int mz = mp * 16, mg = mz + 64;
    v8f az, ag;
#pragma unroll
    for (int r = 0; r < 8; ++r) {
      int Mz = mz + r + 8 * lg, Mg = mg + r + 8 * lg;
      az[r] = pb_[OFF_B2 + Mz] + pb_[OFF_BS + Mz];
      ag[r] = pb_[OFF_B2 + Mg] + pb_[OFF_BS + Mg];
    }
#pragma unroll
    for (int kk = 0; kk < 6; ++kk) {  // Ws @ z
      v16h bb = load_b_frag(&zt[0][0], ZLD, ni * 16, kk * 32, lane);
      v16h a1 = load_a_frag(Wsb, FIN, mz, kk * 32, lane);
      az = wmma_f16(a1, bb, az);
      v16h a2 = load_a_frag(Wsb, FIN, mg, kk * 32, lane);
      ag = wmma_f16(a2, bb, ag);
    }
#pragma unroll
    for (int kk = 0; kk < 2; ++kk) {  // W2 @ h
      v16h bb = load_b_frag(&ht[0][0], HLD, ni * 16, kk * 32, lane);
      v16h a1 = load_a_frag(W2b, FH, mz, kk * 32, lane);
      az = wmma_f16(a1, bb, az);
      v16h a2 = load_a_frag(W2b, FH, mg, kk * 32, lane);
      ag = wmma_f16(a2, bb, ag);
    }
    int pixg = pb * 64 + ni * 16 + (lane & 15);
#pragma unroll
    for (int r = 0; r < 8; ++r) {
      int chn = mz + r + 8 * lg;
      float sig = 1.f / (1.f + __expf(-ag[r]));
      size_t idx = (size_t)(b * NF + chn) * HW + pixg;
      outp[idx] += lf * az[r] * sig;
    }
  }
}

// ---------------- final head: relu-MLP residual + 3-ch projection ------------------

__global__ void __launch_bounds__(256)
final_kernel(const float* __restrict__ out, const float* __restrict__ r1w,
             const float* __restrict__ r1b, const float* __restrict__ r2w,
             const float* __restrict__ r2b, const float* __restrict__ ocw,
             const float* __restrict__ ocb, float* __restrict__ dout, int HW) {
  __shared__ float R1[64 * 64], R2[64 * 64], OC[3 * 64], Bb1[64], Bb2[64], Bb3[3];
  for (int i = threadIdx.x; i < 4096; i += 256) { R1[i] = r1w[i]; R2[i] = r2w[i]; }
  for (int i = threadIdx.x; i < 192; i += 256) OC[i] = ocw[i];
  if (threadIdx.x < 64) { Bb1[threadIdx.x] = r1b[threadIdx.x]; Bb2[threadIdx.x] = r2b[threadIdx.x]; }
  if (threadIdx.x < 3) Bb3[threadIdx.x] = ocb[threadIdx.x];
  __syncthreads();
  int gp = blockIdx.x * 256 + threadIdx.x;
  if (gp >= BSZ * HW) return;
  int b = gp / HW, pp = gp - b * HW;
  const float* ob = out + (size_t)b * NF * HW + pp;
  float o[64], h[64];
#pragma unroll
  for (int c = 0; c < 64; ++c) o[c] = ob[(size_t)c * HW];
  for (int m = 0; m < 64; ++m) {
    float acc = Bb1[m];
    const float* row = &R1[m * 64];
#pragma unroll 8
    for (int i = 0; i < 64; ++i) acc += row[i] * o[i];
    h[m] = acc > 0.f ? acc : 0.f;
  }
  float y0 = Bb3[0], y1 = Bb3[1], y2 = Bb3[2];
  for (int m = 0; m < 64; ++m) {
    float acc = Bb2[m];
    const float* row = &R2[m * 64];
#pragma unroll 8
    for (int i = 0; i < 64; ++i) acc += row[i] * h[i];
    float tm = o[m] + acc;
    y0 += OC[m] * tm;
    y1 += OC[64 + m] * tm;
    y2 += OC[128 + m] * tm;
  }
  size_t half = (size_t)BSZ * 3 * HW;
  float ys[3] = {y0, y1, y2};
#pragma unroll
  for (int oc = 0; oc < 3; ++oc) {
    size_t oidx = (size_t)(b * 3 + oc) * HW + pp;
    dout[oidx] = fminf(fmaxf(ys[oc], -1.f), 1.f);
    dout[half + oidx] = ys[oc];
  }
}

// ---------------- host orchestration ----------------

extern "C" void kernel_launch(void* const* d_in, const int* in_sizes, int n_in,
                              void* d_out, int out_size, void* d_ws, size_t ws_size,
                              hipStream_t stream) {
  const float* lat      = (const float*)d_in[0];
  const float* ca_noise = (const float*)d_in[1];
  // d_in[2] = seed: inst_norm(broadcast(seed)) over HxW has variance 0 -> exactly 0
  const float* lfp      = (const float*)d_in[3];
  const float* hyper_w  = (const float*)d_in[4];
  const float* hyper_b  = (const float*)d_in[5];
  const float* r1w = (const float*)d_in[6];
  const float* r1b = (const float*)d_in[7];
  const float* r2w = (const float*)d_in[8];
  const float* r2b = (const float*)d_in[9];
  const float* ocw = (const float*)d_in[10];
  const float* ocb = (const float*)d_in[11];
  float* dout = (float*)d_out;

  char* base = (char*)d_ws;
  size_t off = 0;
  auto take = [&](size_t bytes) -> char* {
    char* q = base + off;
    off += (bytes + 255) & ~(size_t)255;
    return q;
  };
  float*    pbuf  = (float*)take((size_t)BSZ * PT * 4);
  _Float16* w1h   = (_Float16*)take((size_t)BSZ * FH * FIN * 2);
  _Float16* w2h   = (_Float16*)take((size_t)BSZ * FOUT * FH * 2);
  _Float16* wsh   = (_Float16*)take((size_t)BSZ * FOUT * FIN * 2);
  float*    meanp = (float*)take((size_t)BSZ * FIN * 4);
  float*    rstdp = (float*)take((size_t)BSZ * FIN * 4);
  float*    outA  = (float*)take((size_t)BSZ * NF * 128 * 128 * 4);
  float*    tmp   = (float*)take((size_t)BSZ * NF * 128 * 128 * 4);
  _Float16* zbuf  = (_Float16*)take((size_t)BSZ * 128 * 128 * FIN * 2);
  (void)ws_size; (void)in_sizes; (void)n_in; (void)out_size;

  // hypernetwork params (lat is constant across layers -> compute once)
  hyper_kernel<<<(PT + 255) / 256, 256, 0, stream>>>(lat, hyper_w, hyper_b, pbuf);
  convert_kernel<<<(BSZ * 45056 + 255) / 256, 256, 0, stream>>>(pbuf, w1h, w2h, wsh);

  // out0 = gauss_blur(ca_noise)  (inst_norm(seed broadcast) == 0)
  blur_kernel<<<(BSZ * NF * 16 * 16 + 255) / 256, 256, 0, stream>>>(
      ca_noise, outA, BSZ * NF, 16, 16);

  int H = 16;
  for (int c = 0; c < 16; ++c) {
    int HW = H * H;
    stats_kernel<<<dim3(NF, BSZ), 256, 0, stream>>>(outA, H, H, meanp, rstdp);
    zwrite_kernel<<<dim3(NF, BSZ), 256, 0, stream>>>(outA, meanp, rstdp, zbuf, H, H);
    dyna_kernel<<<dim3(HW / 64, BSZ), 256, 0, stream>>>(zbuf, w1h, w2h, wsh, pbuf,
                                                        lfp, outA, HW);
    if ((c & 3) == 3 && c < 15) {
      int H2 = 2 * H;
      up2_kernel<<<(BSZ * NF * H2 * H2 + 255) / 256, 256, 0, stream>>>(outA, tmp, H, H);
      blur_kernel<<<(BSZ * NF * H2 * H2 + 255) / 256, 256, 0, stream>>>(
          tmp, outA, BSZ * NF, H2, H2);
      H = H2;
    }
  }
  final_kernel<<<(BSZ * 128 * 128 + 255) / 256, 256, 0, stream>>>(
      outA, r1w, r1b, r2w, r2b, ocw, ocb, dout, 128 * 128);
}
